// Encoder_34943853920780
// MI455X (gfx1250) — matrix operand order repaired and checked by reading, not compiled, not run on hardware
//
#include <hip/hip_runtime.h>
#include <hip/hip_bf16.h>
#include <stdint.h>

typedef __attribute__((ext_vector_type(16))) _Float16     v16h;
typedef __attribute__((ext_vector_type(8)))  float        v8f;
typedef __attribute__((ext_vector_type(4)))  unsigned int u32x4;
typedef __attribute__((ext_vector_type(8)))  unsigned int u32x8;

#define DC   2048            // degree cap for stable-rank bucketing
#define CHN  512             // nodes per chunk for stable within-bucket rank
#define HASH_LOG 18
#define HCAP (1u << HASH_LOG)
#define EPB  512             // edges per batch in accumulation kernel

// ---------------- init: zero/fill all workspace state + output ----------------
// Also fills the iab padding region [E, Epad) of both rows with padVal so the
// TDM tile load is always fully in-bounds and tail K-slots hit the zero row.
__global__ void k_init(int* degree, int N, int* chunkHist, int chCount,
                       int* base, unsigned long long* hash, float* out, int D,
                       int* iab, int E, int Epad, int padVal) {
  int stride = gridDim.x * blockDim.x;
  int g = blockIdx.x * blockDim.x + threadIdx.x;
  for (int i = g; i < N; i += stride) degree[i] = 0;
  for (int i = g; i < chCount; i += stride) chunkHist[i] = 0;
  for (int i = g; i < DC; i += stride) base[i] = 0;
  for (int i = g; i < (int)HCAP; i += stride) hash[i] = ~0ull;
  for (int i = g; i < D; i += stride) out[i] = 0.0f;
  for (int i = E + g; i < Epad; i += stride) { iab[i] = padVal; iab[Epad + i] = padVal; }
}

// ---- one-time f32 -> f16 conversion of the weight table, plus a zero row at N ----
__global__ void k_cvt(const float* __restrict__ w, _Float16* __restrict__ wh,
                      unsigned w4, unsigned total4) {  // w4 = N*D/4, total4 = (N+1)*D/4
  unsigned stride = gridDim.x * blockDim.x;
  for (unsigned i = blockIdx.x * blockDim.x + threadIdx.x; i < total4; i += stride) {
    float4 f = (i < w4) ? ((const float4*)w)[i] : make_float4(0.f, 0.f, 0.f, 0.f);
    _Float16* p = wh + (size_t)i * 4;
    p[0] = (_Float16)f.x; p[1] = (_Float16)f.y;
    p[2] = (_Float16)f.z; p[3] = (_Float16)f.w;
  }
}

// ---------------- in-degree over columns ----------------
__global__ void k_degree(const int* __restrict__ col, int* degree, int E, int N) {
  int e = blockIdx.x * blockDim.x + threadIdx.x;
  if (e < E) {
    int c = col[e];
    if (c >= 0 && c < N) atomicAdd(&degree[c], 1);
  }
}

// ---------------- per-chunk degree histograms ----------------
__global__ void k_chunkhist(const int* __restrict__ degree, int* chunkHist, int N) {
  int v = blockIdx.x * blockDim.x + threadIdx.x;
  if (v < N) {
    int d = degree[v]; if (d > DC - 1) d = DC - 1;
    atomicAdd(&chunkHist[(v / CHN) * DC + d], 1);
  }
}

// ---------------- per-degree exclusive prefix over chunks; totals -> base ----------------
__global__ void k_colprefix(int* chunkHist, int* base, int nch) {
  int d = blockIdx.x * blockDim.x + threadIdx.x;
  if (d < DC) {
    int run = 0;
    for (int j = 0; j < nch; ++j) {
      int idx = j * DC + d;
      int t = chunkHist[idx];
      chunkHist[idx] = run;
      run += t;
    }
    base[d] = run;
  }
}

// ---------------- exclusive scan of degree totals (tiny, single thread) ----------------
__global__ void k_scan(int* base) {
  if (blockIdx.x == 0 && threadIdx.x == 0) {
    int ex = 0;
    for (int d = 0; d < DC; ++d) { int t = base[d]; base[d] = ex; ex += t; }
  }
}

// ---------------- stable rank: base[deg] + chunk prefix + in-chunk stable offset ----------------
__global__ void k_rank(const int* __restrict__ degree, const int* __restrict__ chunkHist,
                       const int* __restrict__ base, int* __restrict__ rank, int N) {
  __shared__ int sdeg[CHN];
  int chunk = blockIdx.x;
  int v = chunk * CHN + (int)threadIdx.x;
  int d = -1;
  if (v < N) { d = degree[v]; if (d > DC - 1) d = DC - 1; }
  sdeg[threadIdx.x] = d;
  __syncthreads();
  if (v < N) {
    int cnt = 0;
    for (int u = 0; u < (int)threadIdx.x; ++u) cnt += (sdeg[u] == d);
    rank[v] = base[d] + chunkHist[chunk * DC + d] + cnt;
  }
}

__device__ __forceinline__ unsigned long long mix64(unsigned long long x) {
  x += 0x9E3779B97F4A7C15ull;
  x = (x ^ (x >> 30)) * 0xBF58476D1CE4E5B9ull;
  x = (x ^ (x >> 27)) * 0x94D049BB133111EBull;
  return x ^ (x >> 31);
}

// -- canonicalize + hash-dedup; emit (rank+bias | invalidVal, rank+bias) per edge --
// ZROW encoding: bias=0, invalid=N (zero row).  Fallback: bias=1, invalid=0 (mask).
__global__ void k_dedup(const int* __restrict__ row, const int* __restrict__ col,
                        unsigned long long* hash, const int* __restrict__ rank,
                        int* __restrict__ iab, int E, int Epad, int N,
                        int bias, int invalidVal) {
  int e = blockIdx.x * blockDim.x + threadIdx.x;
  if (e >= E) return;
  int u = row[e], v = col[e];
  if (u < 0) u = 0; if (u >= N) u = N - 1;
  if (v < 0) v = 0; if (v >= N) v = N - 1;
  int rr = min(u, v), cc = max(u, v);
  unsigned long long key = ((unsigned long long)(unsigned)rr << 32) | (unsigned)cc;
  unsigned slot = (unsigned)(mix64(key) & (unsigned long long)(HCAP - 1));
  bool keep = false;
  for (;;) {
    unsigned long long cur = hash[slot];
    if (cur == key) { keep = false; break; }
    if (cur == ~0ull) {
      unsigned long long prev = atomicCAS(&hash[slot], ~0ull, key);
      if (prev == ~0ull) { keep = true; break; }
      if (prev == key)   { keep = false; break; }
    }
    slot = (slot + 1) & (HCAP - 1);
  }
  iab[e]        = keep ? (rank[rr] + bias) : invalidVal;
  iab[Epad + e] = rank[cc] + bias;
}

// ---------------- main accumulation: WMMA f16 16x16x32, diagonal of A^T B ----------------
// grid = (D/128, Epad/EPB), block = 256 (8 waves). Wave w handles dim-tile
// blockIdx.x*8 + w (16 dims), iterating 32-edge groups of this batch, C resident.
// ZROW: stored values are row indices in [0,N], N = zero row => no masks at all.
// LOGD >= 0: D == 1<<LOGD, offsets become a single v_lshl_add_u32.
template <typename T, bool ZROW, int LOGD>
__global__ void k_accum(const T* __restrict__ table, const int* __restrict__ iab,
                        float* __restrict__ out, int Epad, int D, int N) {
  __shared__ int sIab[2 * EPB];
  const int batch = blockIdx.y;
  const int e0 = batch * EPB;

  // --- TDM async tensor load: 2D tile [EPB x 2] of iab (row stride Epad) -> LDS.
  // iab is padded to Epad, so the tile is always fully in-bounds.
  if (threadIdx.x < 32) {
    unsigned long long ga = (unsigned long long)(uintptr_t)(iab + e0);
    unsigned int lds = (unsigned int)(uintptr_t)(void*)&sIab[0];
    unsigned int td0 = (unsigned int)(Epad - e0);
    u32x4 g0;
    g0[0] = 1u;                                     // count=1 (valid user D#)
    g0[1] = lds;                                    // lds_addr
    g0[2] = (unsigned int)ga;                       // global_addr[31:0]
    g0[3] = (unsigned int)(ga >> 32) | (2u << 30);  // global_addr[56:32] | type=2
    u32x8 g1;
    g1[0] = (2u << 16);                             // data_size=4B; no mcast/pad/iter
    g1[1] = (td0 & 0xFFFFu) << 16;                  // tensor_dim0[15:0]
    g1[2] = (td0 >> 16) | (2u << 16);               // tensor_dim0[31:16] | tensor_dim1=2
    g1[3] = ((unsigned)EPB << 16);                  // tile_dim0 = EPB
    g1[4] = 2u;                                     // tile_dim1 = 2, tile_dim2 = 0
    g1[5] = (unsigned int)Epad;                     // tensor_dim0_stride lo
    g1[6] = 0u;                                     // stride0 hi | stride1 lo
    g1[7] = 0u;
    asm volatile("tensor_load_to_lds %0, %1" :: "s"(g0), "s"(g1) : "memory");
    __builtin_amdgcn_s_wait_tensorcnt(0);
  }
  __syncthreads();

  const int lane   = (int)threadIdx.x & 31;
  // wave-uniform by construction; readfirstlane forces SGPR so the weight-table
  // base becomes scalar and gathers lower to saddr+voffset global_load form.
  const int tile   = __builtin_amdgcn_readfirstlane(blockIdx.x * 8 +
                                                    ((int)threadIdx.x >> 5));
  const int kh     = lane >> 4;                     // lane half (K offset select)
  const unsigned m = (unsigned)(lane & 15);         // M (A) / N (B) index = dim
  const unsigned Du = (unsigned)D;

  if (tile * 16 < D) {
    const char* wbase = (const char*)(table + tile * 16);   // SGPR base
    v8f c = {};
#pragma unroll 2
    for (int g = 0; g < EPB / 32; ++g) {
      const int gbase = g * 32;
      v16h a, b;
#pragma unroll
      for (int r = 0; r < 8; ++r) {
        // A 16x32 f16, this code's k order (not the ISA table): vgpr r: K = 16h + 2r (pairs), matching B
        int ka = 16 * kh + 2 * r;
        int2 va = *(const int2*)&sIab[gbase + ka];            // merged ds loads
        // B 32x16 f16, same k order as A (not the ISA table): vgpr j: K = 16h + 2j (pairs), N = lane&15
        int kb = 16 * kh + 2 * r;
        int2 vb = *(const int2*)&sIab[EPB + gbase + kb];
        if (ZROW) {
          // values already in [0,N]; N is the zero row: 1 lshl_add + 1 load each
          unsigned o0, o1, o2, o3;
          if (LOGD >= 0) {
            o0 = (((unsigned)va.x << LOGD) + m) * (unsigned)sizeof(T);
            o1 = (((unsigned)va.y << LOGD) + m) * (unsigned)sizeof(T);
            o2 = (((unsigned)vb.x << LOGD) + m) * (unsigned)sizeof(T);
            o3 = (((unsigned)vb.y << LOGD) + m) * (unsigned)sizeof(T);
          } else {
            o0 = ((unsigned)va.x * Du + m) * (unsigned)sizeof(T);
            o1 = ((unsigned)va.y * Du + m) * (unsigned)sizeof(T);
            o2 = ((unsigned)vb.x * Du + m) * (unsigned)sizeof(T);
            o3 = ((unsigned)vb.y * Du + m) * (unsigned)sizeof(T);
          }
          a[2 * r]     = (_Float16)*(const T*)(wbase + o0);
          a[2 * r + 1] = (_Float16)*(const T*)(wbase + o1);
          b[2 * r]     = (_Float16)*(const T*)(wbase + o2);
          b[2 * r + 1] = (_Float16)*(const T*)(wbase + o3);
        } else {
          unsigned r0 = (unsigned)max(va.x - 1, 0);
          unsigned r1 = (unsigned)max(va.y - 1, 0);
          unsigned r2 = (unsigned)max(vb.x - 1, 0);
          unsigned r3 = (unsigned)max(vb.y - 1, 0);
          _Float16 a0 = (_Float16)*(const T*)(wbase + (r0 * Du + m) * (unsigned)sizeof(T));
          _Float16 a1 = (_Float16)*(const T*)(wbase + (r1 * Du + m) * (unsigned)sizeof(T));
          a[2 * r]     = (va.x != 0) ? a0 : (_Float16)0.0f;
          a[2 * r + 1] = (va.y != 0) ? a1 : (_Float16)0.0f;
          b[2 * r]     = (_Float16)*(const T*)(wbase + (r2 * Du + m) * (unsigned)sizeof(T));
          b[2 * r + 1] = (_Float16)*(const T*)(wbase + (r3 * Du + m) * (unsigned)sizeof(T));
        }
      }
      c = __builtin_amdgcn_wmma_f32_16x16x32_f16(false, a, false, b,
                                                 (short)0, c, false, false);
    }
    // diagonal extraction: C[m,m] lives at (vgpr=m, lane=m) for m<8,
    // and (vgpr=m-8, lane=m+16) for m in [8,16)
    int vr = -1, md = -1;
    if (lane < 8)        { vr = lane;      md = lane;      }
    else if (lane >= 24) { vr = lane - 24; md = lane - 16; }
    if (vr >= 0) {
      float dv = c[0];
#pragma unroll
      for (int i = 1; i < 8; ++i) dv = (vr == i) ? c[i] : dv;
      atomicAdd(&out[tile * 16 + md], dv);
    }
  }
}

extern "C" void kernel_launch(void* const* d_in, const int* in_sizes, int n_in,
                              void* d_out, int out_size, void* d_ws, size_t ws_size,
                              hipStream_t stream) {
  const float* w    = (const float*)d_in[0];
  const int*   edge = (const int*)d_in[1];
  const int D = out_size;                 // 1024
  const int N = in_sizes[0] / D;          // 50000
  const int E = in_sizes[1] / 2;          // 100000
  const int nch = (N + CHN - 1) / CHN;
  const int nbatch = (E + EPB - 1) / EPB;
  const int Epad = nbatch * EPB;

  char* ws = (char*)d_ws;
  size_t off = 0;
  auto alloc = [&](size_t bytes) -> char* {
    off = (off + 255) & ~(size_t)255;
    char* p = ws + off;
    off += bytes;
    return p;
  };
  int* degree              = (int*)alloc((size_t)N * 4);
  int* chunkHist           = (int*)alloc((size_t)nch * DC * 4);
  int* base                = (int*)alloc((size_t)DC * 4);
  int* rank                = (int*)alloc((size_t)N * 4);
  unsigned long long* hash = (unsigned long long*)alloc((size_t)HCAP * 8);
  int* iab                 = (int*)alloc((size_t)2 * Epad * 4);
  _Float16* wh             = (_Float16*)alloc((size_t)(N + 1) * D * 2); // + zero row
  const bool use_f16 = (off <= ws_size);           // f16 table fits in workspace?
  float* out = (float*)d_out;

  int logd = -1;
  for (int s = 0; s < 31; ++s) if ((1 << s) == D) { logd = s; break; }

  k_init<<<dim3(512), dim3(256), 0, stream>>>(degree, N, chunkHist, nch * DC,
                                              base, hash, out, D,
                                              iab, E, Epad, use_f16 ? N : 0);
  if (use_f16) {
    unsigned w4     = (unsigned)(((size_t)N * D) / 4);
    unsigned total4 = (unsigned)(((size_t)(N + 1) * D) / 4);
    k_cvt<<<dim3(2048), dim3(256), 0, stream>>>(w, wh, w4, total4);
  }
  k_degree<<<dim3((E + 255) / 256), dim3(256), 0, stream>>>(edge + E, degree, E, N);
  k_chunkhist<<<dim3((N + 255) / 256), dim3(256), 0, stream>>>(degree, chunkHist, N);
  k_colprefix<<<dim3((DC + 255) / 256), dim3(256), 0, stream>>>(chunkHist, base, nch);
  k_scan<<<dim3(1), dim3(32), 0, stream>>>(base);
  k_rank<<<dim3(nch), dim3(CHN), 0, stream>>>(degree, chunkHist, base, rank, N);
  k_dedup<<<dim3((E + 255) / 256), dim3(256), 0, stream>>>(
      edge, edge + E, hash, rank, iab, E, Epad, N,
      use_f16 ? 0 : 1, use_f16 ? N : 0);
  dim3 gAcc((D / 16 + 7) / 8, nbatch);
  if (use_f16) {
    if (logd == 10) {
      k_accum<_Float16, true, 10><<<gAcc, dim3(256), 0, stream>>>(wh, iab, out, Epad, D, N);
    } else {
      k_accum<_Float16, true, -1><<<gAcc, dim3(256), 0, stream>>>(wh, iab, out, Epad, D, N);
    }
  } else {
    k_accum<float, false, -1><<<gAcc, dim3(256), 0, stream>>>(w, iab, out, Epad, D, N);
  }
}
